// CustomModel_60851096649964
// MI455X (gfx1250) — compile-verified, matrix operand order checked by reading, not run on hardware
//
#include <hip/hip_runtime.h>
#include <stdint.h>

#define NHEADS 8
#define DIM 128
#define A_TOK 10
#define M_TOK 350
#define TK_SELF 3
#define TK_CROSS 32
#define BT 2880

typedef __attribute__((ext_vector_type(16))) _Float16 v16h;
typedef __attribute__((ext_vector_type(8)))  float    v8f;

union FragH { v16h v; _Float16 e[16]; };
union FragF { v8f  v; float    e[8];  };

__device__ __forceinline__ int kmap(int g, int i) {
  // 16-bit A/B fragment K index pattern (ISA 7.12.2): lane-half g selects K block
  return g * 8 + (i & 7) + ((i >= 8) ? 16 : 0);
}

__device__ __forceinline__ v8f wmma16(v16h a, v16h b, v8f c) {
  return __builtin_amdgcn_wmma_f32_16x16x32_f16(false, a, false, b, (short)0, c,
                                                false, false);
}

// async copy of 16 bytes global -> LDS (ASYNCcnt tracked, bypasses VGPRs)
__device__ __forceinline__ void async_g2l_b128(uint32_t lds_off,
                                               const void* gptr) {
  asm volatile("global_load_async_to_lds_b128 %0, %1, off"
               :
               : "v"(lds_off), "v"(gptr)
               : "memory");
}

__device__ __forceinline__ void wait_async0() {
  asm volatile("s_wait_asynccnt 0" ::: "memory");
}

__device__ __forceinline__ uint32_t lds_off_of(const void* p) {
  // generic LDS pointer: high 32 bits = shared aperture, low 32 = LDS offset
  return (uint32_t)(uintptr_t)p;
}

// ---------------------------------------------------------------------------
// Kernel 1: fused cross attention.  One workgroup per batch element b.
//   q = a_pad(16x128) @ Wq                        (WMMA)
//   stream m in 16-row tiles (async double-buffer): k,v = m @ Wk,Wv  (WMMA)
//   per-head biased online softmax + PV           (VALU, tiny FLOPs)
//   y = attn_out(16x128) @ Wcross -> ws (f16)     (WMMA)
// ---------------------------------------------------------------------------
__global__ void __launch_bounds__(256)
cross_attn_kernel(const float* __restrict__ a_token,
                  const float* __restrict__ m_token,
                  const float* __restrict__ a2m_pe,
                  const float* __restrict__ Wq,
                  const float* __restrict__ Wk,
                  const float* __restrict__ Wv,
                  const float* __restrict__ Wcross,
                  const int*   __restrict__ a2m_rel,
                  _Float16* __restrict__ y_out) // [BT][16][128] f16
{
  const int b    = blockIdx.x;
  const int tid  = threadIdx.x;
  const int wave = tid >> 5;     // 8 waves: N-tile in GEMMs, head in attention
  const int lane = tid & 31;
  const int ln   = lane & 15;
  const int g    = lane >> 4;
  const int h    = wave;
  const int wt   = wave;

  __shared__ float    m_sf[2][16][DIM]; // async-filled double buffer (f32)
  __shared__ _Float16 a_s[16][DIM];
  __shared__ float    q_s[16][DIM];
  __shared__ float    k_s[16][DIM];
  __shared__ float    v_s[16][DIM];
  __shared__ float    ao_s[16][DIM];
  __shared__ int      rel_s[16][16];

  const int NT = (M_TOK + 15) / 16; // 22 tiles

  // ---- issue async load of m tile 0 (overlaps q computation) ----
  {
#pragma unroll
    for (int qq = 0; qq < 2; ++qq) {
      int slot = tid + qq * 256;      // 512 float4 slots
      int r = slot >> 5;              // 0..15
      int c4 = (slot & 31) * 4;       // 0..124
      int mr = r;                     // tile 0
      if (mr > M_TOK - 1) mr = M_TOK - 1;
      async_g2l_b128(lds_off_of(&m_sf[0][r][c4]),
                     m_token + ((size_t)b * M_TOK + mr) * DIM + c4);
    }
  }

  // ---- stage a_token (pad 10 -> 16 rows), f32 -> f16 ----
  for (int t = tid; t < 16 * DIM; t += 256) {
    int r = t >> 7, c = t & 127;
    float v = (r < A_TOK) ? a_token[((size_t)b * A_TOK + r) * DIM + c] : 0.f;
    a_s[r][c] = (_Float16)v;
  }
  __syncthreads();

  // ---- q = a @ Wq (each wave: columns [wt*16, wt*16+16)) ----
  {
    FragF acc;
#pragma unroll
    for (int i = 0; i < 8; ++i) acc.e[i] = 0.f;
#pragma unroll
    for (int ks = 0; ks < 4; ++ks) {
      FragH af, bf;
#pragma unroll
      for (int i = 0; i < 16; ++i) {
        int kk = ks * 32 + kmap(g, i);
        af.e[i] = a_s[ln][kk];
        bf.e[i] = (_Float16)Wq[(size_t)kk * DIM + wt * 16 + ln];
      }
      acc.v = wmma16(af.v, bf.v, acc.v);
    }
#pragma unroll
    for (int r = 0; r < 8; ++r)
      q_s[r + 8 * g][wt * 16 + ln] = acc.e[r];
  }
  __syncthreads();

  // per-head query registers (agent = ln, head = wave)
  float q_reg[16];
#pragma unroll
  for (int d = 0; d < 16; ++d) q_reg[d] = q_s[ln][h * 16 + d];

  // ---- preload Wk / Wv B-fragments (register resident across m loop) ----
  v16h bk[4], bv[4];
#pragma unroll
  for (int ks = 0; ks < 4; ++ks) {
    FragH fk, fv;
#pragma unroll
    for (int i = 0; i < 16; ++i) {
      int kk = ks * 32 + kmap(g, i);
      fk.e[i] = (_Float16)Wk[(size_t)kk * DIM + wt * 16 + ln];
      fv.e[i] = (_Float16)Wv[(size_t)kk * DIM + wt * 16 + ln];
    }
    bk[ks] = fk.v;
    bv[ks] = fv.v;
  }

  float o[16];
#pragma unroll
  for (int d = 0; d < 16; ++d) o[d] = 0.f;
  float m_run = -__builtin_inff();
  float l_run = 0.f;

  for (int mt = 0; mt < NT; ++mt) {
    const int buf = mt & 1;

    // tile mt has landed in LDS; also orders prior score phase before
    // overwriting rel_s / k_s / v_s / issuing into buf^1
    wait_async0();
    __syncthreads();

    // issue async load of next tile into the other buffer
    if (mt + 1 < NT) {
#pragma unroll
      for (int qq = 0; qq < 2; ++qq) {
        int slot = tid + qq * 256;
        int r = slot >> 5;
        int c4 = (slot & 31) * 4;
        int mr = (mt + 1) * 16 + r;
        if (mr > M_TOK - 1) mr = M_TOK - 1; // clamped rows are rel-masked
        async_g2l_b128(lds_off_of(&m_sf[buf ^ 1][r][c4]),
                       m_token + ((size_t)b * M_TOK + mr) * DIM + c4);
      }
    }

    // relation tile
    {
      int aa = tid >> 4, j = tid & 15;
      int mr = mt * 16 + j;
      rel_s[aa][j] = (aa < A_TOK && mr < M_TOK)
                         ? a2m_rel[((size_t)b * A_TOK + aa) * M_TOK + mr]
                         : -1;
    }

    // k,v projection tiles via WMMA (A fragments: f32 LDS -> f16)
    {
      FragF ck, cv;
#pragma unroll
      for (int i = 0; i < 8; ++i) { ck.e[i] = 0.f; cv.e[i] = 0.f; }
#pragma unroll
      for (int ks = 0; ks < 4; ++ks) {
        FragH af;
#pragma unroll
        for (int i = 0; i < 16; ++i)
          af.e[i] = (_Float16)m_sf[buf][ln][ks * 32 + kmap(g, i)];
        ck.v = wmma16(af.v, bk[ks], ck.v);
        cv.v = wmma16(af.v, bv[ks], cv.v);
      }
#pragma unroll
      for (int r = 0; r < 8; ++r) {
        k_s[r + 8 * g][wt * 16 + ln] = ck.e[r];
        v_s[r + 8 * g][wt * 16 + ln] = cv.e[r];
      }
    }
    __syncthreads();

    // biased scores + online softmax, head h; lane = (agent ln, row-half g)
    float s[8];
#pragma unroll
    for (int jj = 0; jj < 8; ++jj) {
      int j = g * 8 + jj;
      float acc = 0.f;
#pragma unroll
      for (int d = 0; d < 16; ++d) acc += q_reg[d] * k_s[j][h * 16 + d];
      acc *= 0.25f; // 1/sqrt(16)
      int rel = rel_s[ln][j];
      if (rel >= 0)
        acc += a2m_pe[(((size_t)b * A_TOK + ln) * TK_CROSS + rel) * NHEADS + h];
      else
        acc = -__builtin_inff();
      s[jj] = acc;
    }
    float tmax = s[0];
#pragma unroll
    for (int jj = 1; jj < 8; ++jj) tmax = fmaxf(tmax, s[jj]);
    tmax = fmaxf(tmax, __shfl_xor(tmax, 16, 32));
    float new_m = fmaxf(m_run, tmax);
    if (new_m != -__builtin_inff()) {
      float scale = __expf(m_run - new_m); // exp(-inf)=0 on first hit
      float p[8], lsum = 0.f;
#pragma unroll
      for (int jj = 0; jj < 8; ++jj) {
        p[jj] = __expf(s[jj] - new_m); // masked rows: exp(-inf)=0
        lsum += p[jj];
      }
      l_run = l_run * scale + lsum + __shfl_xor(lsum, 16, 32);
#pragma unroll
      for (int d = 0; d < 16; ++d) {
        float part = 0.f;
#pragma unroll
        for (int jj = 0; jj < 8; ++jj)
          part += p[jj] * v_s[g * 8 + jj][h * 16 + d];
        o[d] = o[d] * scale + part + __shfl_xor(part, 16, 32);
      }
      m_run = new_m;
    }
  }

  // normalize and stash per-head output
  float inv = (l_run > 0.f) ? (1.f / l_run) : 0.f;
  if (g == 0) {
#pragma unroll
    for (int d = 0; d < 16; ++d) ao_s[ln][h * 16 + d] = o[d] * inv;
  }
  __syncthreads();

  // y = ao @ Wcross -> workspace (f16, padded 16 rows)
  {
    FragF acc;
#pragma unroll
    for (int i = 0; i < 8; ++i) acc.e[i] = 0.f;
#pragma unroll
    for (int ks = 0; ks < 4; ++ks) {
      FragH af, bf;
#pragma unroll
      for (int i = 0; i < 16; ++i) {
        int kk = ks * 32 + kmap(g, i);
        af.e[i] = (_Float16)ao_s[ln][kk];
        bf.e[i] = (_Float16)Wcross[(size_t)kk * DIM + wt * 16 + ln];
      }
      acc.v = wmma16(af.v, bf.v, acc.v);
    }
#pragma unroll
    for (int r = 0; r < 8; ++r)
      y_out[((size_t)b * 16 + (r + 8 * g)) * DIM + wt * 16 + ln] =
          (_Float16)acc.e[r];
  }
}

// ---------------------------------------------------------------------------
// Kernel 2: fused self attention.  One workgroup per batch element.
//   qkv = y(16x128) @ W_cattn(128x384)   (WMMA, 3 regions; y staged async)
//   per-head biased softmax over 10 keys (VALU)
//   out = ao(16x128) @ W_self            (WMMA), rows 0..9 -> d_out
// ---------------------------------------------------------------------------
__global__ void __launch_bounds__(256)
self_attn_kernel(const _Float16* __restrict__ y_in, // [BT][16][128] f16
                 const float* __restrict__ a_pe,    // [BT][10][3][8]
                 const float* __restrict__ Wcattn,  // [128][384]
                 const float* __restrict__ Wself,   // [128][128]
                 const int*   __restrict__ a_rel,   // [BT][10][10]
                 float* __restrict__ out)           // [BT][10][128]
{
  const int b    = blockIdx.x;
  const int tid  = threadIdx.x;
  const int wave = tid >> 5;
  const int lane = tid & 31;
  const int ln   = lane & 15;
  const int g    = lane >> 4;
  const int h    = wave;
  const int wt   = wave;

  __shared__ _Float16 y_s[16][DIM];
  __shared__ float    q2[16][DIM];
  __shared__ float    k2[16][DIM];
  __shared__ float    v2[16][DIM];
  __shared__ float    ao[16][DIM];

  // stage y tile via async LDS DMA: 256 x b128 (8 halves each)
  {
    int r = tid >> 4;          // 0..15
    int c8 = (tid & 15) * 8;   // 0..120
    async_g2l_b128(lds_off_of(&y_s[r][c8]),
                   y_in + ((size_t)b * 16 + r) * DIM + c8);
  }
  wait_async0();
  __syncthreads();

  // qkv projection: region rg in {q,k,v}, wave handles 16 columns of each
#pragma unroll
  for (int rg = 0; rg < 3; ++rg) {
    FragF acc;
#pragma unroll
    for (int i = 0; i < 8; ++i) acc.e[i] = 0.f;
#pragma unroll
    for (int ks = 0; ks < 4; ++ks) {
      FragH af, bf;
#pragma unroll
      for (int i = 0; i < 16; ++i) {
        int kk = ks * 32 + kmap(g, i);
        af.e[i] = y_s[ln][kk];
        bf.e[i] =
            (_Float16)Wcattn[(size_t)kk * (3 * DIM) + rg * DIM + wt * 16 + ln];
      }
      acc.v = wmma16(af.v, bf.v, acc.v);
    }
    float(*dst)[DIM] = (rg == 0) ? q2 : ((rg == 1) ? k2 : v2);
#pragma unroll
    for (int r = 0; r < 8; ++r)
      dst[r + 8 * g][wt * 16 + ln] = acc.e[r];
  }
  __syncthreads();

  // self attention: head h, agent ln, key-half g (keys j = g*8 + jj)
  float q_reg[16];
#pragma unroll
  for (int d = 0; d < 16; ++d) q_reg[d] = q2[ln][h * 16 + d];

  float s[8];
#pragma unroll
  for (int jj = 0; jj < 8; ++jj) {
    int j = g * 8 + jj;
    float acc = 0.f;
#pragma unroll
    for (int d = 0; d < 16; ++d) acc += q_reg[d] * k2[j][h * 16 + d];
    acc *= 0.25f;
    int rel = (ln < A_TOK && j < A_TOK)
                  ? a_rel[((size_t)b * A_TOK + ln) * A_TOK + j]
                  : -1;
    if (rel >= 0)
      acc += a_pe[(((size_t)b * A_TOK + ln) * TK_SELF + rel) * NHEADS + h];
    else
      acc = -__builtin_inff();
    s[jj] = acc;
  }
  float mx = s[0];
#pragma unroll
  for (int jj = 1; jj < 8; ++jj) mx = fmaxf(mx, s[jj]);
  mx = fmaxf(mx, __shfl_xor(mx, 16, 32));
  float p[8], lsum = 0.f;
  if (mx != -__builtin_inff()) {
#pragma unroll
    for (int jj = 0; jj < 8; ++jj) {
      p[jj] = __expf(s[jj] - mx);
      lsum += p[jj];
    }
  } else {
#pragma unroll
    for (int jj = 0; jj < 8; ++jj) p[jj] = 0.f;
  }
  float l = lsum + __shfl_xor(lsum, 16, 32);
  float inv = (l > 0.f) ? (1.f / l) : 0.f;
#pragma unroll
  for (int d = 0; d < 16; ++d) {
    float part = 0.f;
#pragma unroll
    for (int jj = 0; jj < 8; ++jj) part += p[jj] * v2[g * 8 + jj][h * 16 + d];
    float od = (part + __shfl_xor(part, 16, 32)) * inv;
    if (g == 0) ao[ln][h * 16 + d] = od;
  }
  __syncthreads();

  // final projection: out rows 0..9
  {
    FragF acc;
#pragma unroll
    for (int i = 0; i < 8; ++i) acc.e[i] = 0.f;
#pragma unroll
    for (int ks = 0; ks < 4; ++ks) {
      FragH af, bf;
#pragma unroll
      for (int i = 0; i < 16; ++i) {
        int kk = ks * 32 + kmap(g, i);
        af.e[i] = (_Float16)ao[ln][kk];
        bf.e[i] = (_Float16)Wself[(size_t)kk * DIM + wt * 16 + ln];
      }
      acc.v = wmma16(af.v, bf.v, acc.v);
    }
#pragma unroll
    for (int r = 0; r < 8; ++r) {
      int row = r + 8 * g;
      if (row < A_TOK)
        out[((size_t)b * A_TOK + row) * DIM + wt * 16 + ln] = acc.e[r];
    }
  }
}

// ---------------------------------------------------------------------------
extern "C" void kernel_launch(void* const* d_in, const int* in_sizes, int n_in,
                              void* d_out, int out_size, void* d_ws,
                              size_t ws_size, hipStream_t stream) {
  const float* a_token = (const float*)d_in[0];
  const float* m_token = (const float*)d_in[1];
  const float* a2m_pe  = (const float*)d_in[2];
  const float* a_pe    = (const float*)d_in[3];
  const float* Wq      = (const float*)d_in[4];
  const float* Wk      = (const float*)d_in[5];
  const float* Wv      = (const float*)d_in[6];
  const float* Wcross  = (const float*)d_in[7];
  const float* Wcattn  = (const float*)d_in[8];
  const float* Wself   = (const float*)d_in[9];
  const int* a2m_rel   = (const int*)d_in[10];
  const int* a_rel     = (const int*)d_in[11];

  _Float16* y_ws = (_Float16*)d_ws; // BT*16*128 f16 = 11.8 MB scratch

  cross_attn_kernel<<<BT, 256, 0, stream>>>(a_token, m_token, a2m_pe, Wq, Wk,
                                            Wv, Wcross, a2m_rel, y_ws);
  self_attn_kernel<<<BT, 256, 0, stream>>>(y_ws, a_pe, Wcattn, Wself, a_rel,
                                           (float*)d_out);
}
